// TripletLoss_37976100831762
// MI455X (gfx1250) — matrix operand order repaired and checked by reading, not compiled, not run on hardware
//
#include <hip/hip_runtime.h>

// ---------------------------------------------------------------------------
// Batch-hard triplet loss, N=8192, D=128, f32 in/out.  (MI455X / gfx1250)
//
//   K1: row L2-normalize; emit zn_f32 (exact path), zn_f16 (WMMA path),
//       packed key[i] = label | (subject<<8).
//   K2: fused Gram + mining: v_wmma_f32_16x16x32_f16 (K=128 = 4 chained
//       WMMAs). B panels double-buffered in LDS, filled by
//       GLOBAL_LOAD_ASYNC_TO_LDS_B128 (ASYNCcnt) so DMA overlaps compute.
//       Mining compares the raw Gram value g (rows are unit-norm, so
//       distance is monotone-decreasing in g): hardest positive = min g,
//       hardest negative = max g. Minimal epilogue per pair.
//   K3: exact per-row d_ap/d_an in f32 with PD_EPS, hinge.
//   K4: deterministic fixed-order reduction -> scalar.
// ---------------------------------------------------------------------------

typedef __attribute__((ext_vector_type(16))) _Float16 v16h;
typedef __attribute__((ext_vector_type(8)))  _Float16 v8h;
typedef __attribute__((ext_vector_type(4)))  _Float16 v4h;
typedef __attribute__((ext_vector_type(8)))  float    v8f;

#define N_ROWS 8192
#define DIMS   128
#define PANEL  128
#define LDS_STRIDE_H 136   // halves per LDS row: 128 + 8 pad (272 B row skew)
#define BUF_H (PANEL * LDS_STRIDE_H)

#define MARGIN 0.2f
#define PD_EPS 1e-6f

// ---------------------------------------------------------------------------
// K1: normalize rows -> zn32, zn16, packed key. One wave per row.
// ---------------------------------------------------------------------------
__global__ __launch_bounds__(256)
void tl_normalize_kernel(const float* __restrict__ z,
                         const int* __restrict__ labels,
                         const int* __restrict__ subjects,
                         float* __restrict__ zn32,
                         _Float16* __restrict__ zn16,
                         int* __restrict__ key)
{
    const int wave = threadIdx.x >> 5;
    const int lane = threadIdx.x & 31;
    const int row  = blockIdx.x * 8 + wave;

    const float4 zz = *(const float4*)(z + (size_t)row * DIMS + lane * 4);
    float ss = zz.x*zz.x + zz.y*zz.y + zz.z*zz.z + zz.w*zz.w;
#pragma unroll
    for (int m = 1; m < 32; m <<= 1) ss += __shfl_xor(ss, m, 32);

    const float inv = 1.0f / fmaxf(sqrtf(ss), 1e-12f);

    float4 o;
    o.x = zz.x * inv; o.y = zz.y * inv; o.z = zz.z * inv; o.w = zz.w * inv;
    *(float4*)(zn32 + (size_t)row * DIMS + lane * 4) = o;

    v4h h;
    h[0] = (_Float16)o.x; h[1] = (_Float16)o.y;
    h[2] = (_Float16)o.z; h[3] = (_Float16)o.w;
    *(v4h*)(zn16 + (size_t)row * DIMS + lane * 4) = h;

    if (lane == 0)
        key[row] = (labels[row] & 0xff) | (subjects[row] << 8);
}

// ---------------------------------------------------------------------------
// K2: fused Gram + batch-hard mining with WMMA; double-buffered async LDS.
//   Block: 256 threads (8 waves) covers 128 anchor rows; grid = 64.
// ---------------------------------------------------------------------------
__global__ __launch_bounds__(256)
void tl_mine_kernel(const _Float16* __restrict__ zn16,
                    const int* __restrict__ key,
                    int* __restrict__ pidx,
                    int* __restrict__ nidx)
{
    __shared__ _Float16 ldsB[2][BUF_H];    // 2 x 34816 B
    __shared__ int      ldsKey[2][PANEL];

    const int tid  = threadIdx.x;
    const int wave = tid >> 5;
    const int lane = tid & 31;
    const int half = lane >> 4;     // 0: lanes 0-15, 1: lanes 16-31
    const int l16  = lane & 15;

    const int i0 = blockIdx.x * 128 + wave * 16;

    // 32-bit LDS byte addresses for the async-DMA destinations (low 32 bits
    // of a generic LDS pointer are the LDS offset per the flat-aperture map).
    const unsigned ldsB_base = (unsigned)(size_t)&ldsB[0][0];
    unsigned ldsDst[8];     // per-thread chunk dests in buffer 0
    unsigned gOff[8];       // byte offset of each chunk within a 32 KB panel
#pragma unroll
    for (int c = 0; c < 8; ++c) {
        const int chunk = tid + 256 * c;                 // 2048 x 16 B chunks
        ldsDst[c] = ldsB_base +
                    (unsigned)((chunk >> 4) * (LDS_STRIDE_H * 2) + (chunk & 15) * 16);
        gOff[c]   = (unsigned)(chunk * 16);
    }

    // Issue one panel's async global->LDS DMA (no wait) + stage its keys.
    auto stage_panel = [&](int jp, int buf) {
        const unsigned gbase = (unsigned)(jp * (DIMS * 2));
        const unsigned dbase = (unsigned)(buf * (BUF_H * 2));
#pragma unroll
        for (int c = 0; c < 8; ++c) {
            asm volatile("global_load_async_to_lds_b128 %0, %1, %2"
                         :: "v"(ldsDst[c] + dbase), "v"(gbase + gOff[c]),
                            "s"(zn16)
                         : "memory");
        }
        if (tid < PANEL) ldsKey[buf][tid] = key[jp + tid];
    };

    // --- A tiles (16 rows x K=128) in WMMA 16-bit A layout, kept in VGPRs.
    v16h A[4];
    {
        const _Float16* arow = zn16 + (size_t)(i0 + l16) * DIMS;
#pragma unroll
        for (int kk = 0; kk < 4; ++kk) {
            v8h lo = *(const v8h*)(arow + kk * 32 + half * 8);
            v8h hi = *(const v8h*)(arow + kk * 32 + 16 + half * 8);
            v16h a;
#pragma unroll
            for (int e = 0; e < 8; ++e) { a[e] = lo[e]; a[e + 8] = hi[e]; }
            A[kk] = a;
        }
    }

    // Per-lane metadata for this lane's 8 C-matrix rows: M = v + 8*half.
    int li[8], si[8];
#pragma unroll
    for (int v = 0; v < 8; ++v) {
        const int k = key[i0 + v + half * 8];
        li[v] = k & 0xff;
        si[v] = k >> 8;
    }

    // Running best per (row, n-phase): mined directly on Gram value g.
    //   hardest positive (max dist) == min g ; hardest negative == max g.
    float bestP[8], bestN[8];
    int   bestPi[8], bestNi[8];
#pragma unroll
    for (int v = 0; v < 8; ++v) {
        bestP[v] = 1e30f; bestN[v] = -1e30f; bestPi[v] = -1; bestNi[v] = -1;
    }

    // Prefetch panel 0.
    stage_panel(0, 0);
    asm volatile("s_wait_asynccnt 0x0" ::: "memory");
    __syncthreads();

    int cur = 0;
    for (int jp = 0; jp < N_ROWS; jp += PANEL) {
        // Prefetch next panel into the other buffer (overlaps compute).
        if (jp + PANEL < N_ROWS) stage_panel(jp + PANEL, cur ^ 1);

        for (int sub = 0; sub < PANEL / 16; ++sub) {
            // B tile (K=128 x 16 cols): lane holds column n=l16.
            const _Float16* bbase =
                &ldsB[cur][0] + (sub * 16 + l16) * LDS_STRIDE_H + half * 8;
            v16h B[4];
#pragma unroll
            for (int kk = 0; kk < 4; ++kk) {
                v8h blo = *(const v8h*)(bbase + kk * 32);
                v8h bhi = *(const v8h*)(bbase + kk * 32 + 16);
                v16h b;
#pragma unroll
                for (int e = 0; e < 8; ++e) { b[e] = blo[e]; b[e + 8] = bhi[e]; }
                B[kk] = b;
            }

            v8f c = {};
#pragma unroll
            for (int kk = 0; kk < 4; ++kk)
                c = __builtin_amdgcn_wmma_f32_16x16x32_f16(
                        false, A[kk], false, B[kk], (short)0, c, false, false);

            const int j    = jp + sub * 16 + l16;
            const int keyj = ldsKey[cur][sub * 16 + l16];
            const int lj   = keyj & 0xff;
            const int sj   = keyj >> 8;

#pragma unroll
            for (int v = 0; v < 8; ++v) {
                const float g = c[v];
                const bool sameC = (li[v] == lj);
                const bool sameS = (si[v] == sj);
                const bool pos = sameC && !sameS;   // self fails (!sameS)
                const bool neg = !sameC && sameS;   // self fails (!sameC)
                if (pos && g < bestP[v]) { bestP[v] = g; bestPi[v] = j; }
                if (neg && g > bestN[v]) { bestN[v] = g; bestNi[v] = j; }
            }
        }

        asm volatile("s_wait_asynccnt 0x0" ::: "memory"); // my DMA done
        __syncthreads();                                  // everyone's done
        cur ^= 1;
    }

    // Reduce per-row best across the 16 lanes of each half (xor 1,2,4,8).
    // P: keep smaller g (tie -> smaller index); N: keep larger g.
#pragma unroll
    for (int v = 0; v < 8; ++v) {
        float pv = bestP[v]; int pi = bestPi[v];
        float nv = bestN[v]; int ni = bestNi[v];
#pragma unroll
        for (int m = 1; m < 16; m <<= 1) {
            const float opv = __shfl_xor(pv, m, 32);
            const int   opi = __shfl_xor(pi, m, 32);
            const float onv = __shfl_xor(nv, m, 32);
            const int   oni = __shfl_xor(ni, m, 32);
            if (opv < pv || (opv == pv && opi >= 0 && (pi < 0 || opi < pi))) {
                pv = opv; pi = opi;
            }
            if (onv > nv || (onv == nv && oni >= 0 && (ni < 0 || oni < ni))) {
                nv = onv; ni = oni;
            }
        }
        if (l16 == 0) {
            const int r = i0 + v + half * 8;
            pidx[r] = pi;
            nidx[r] = ni;
        }
    }
}

// ---------------------------------------------------------------------------
// K3: exact per-row hinge (f32 recompute with PD_EPS). One wave per row.
// ---------------------------------------------------------------------------
__global__ __launch_bounds__(256)
void tl_loss_kernel(const float* __restrict__ zn32,
                    const int* __restrict__ pidx,
                    const int* __restrict__ nidx,
                    float* __restrict__ per_row)
{
    const int wave = threadIdx.x >> 5;
    const int lane = threadIdx.x & 31;
    const int row  = blockIdx.x * 8 + wave;

    const int pi = pidx[row], ni = nidx[row];
    const bool valid = (pi >= 0) && (ni >= 0);
    const int ps = valid ? pi : 0;
    const int ns = valid ? ni : 0;

    const float4 a = *(const float4*)(zn32 + (size_t)row * DIMS + lane * 4);
    const float4 p = *(const float4*)(zn32 + (size_t)ps  * DIMS + lane * 4);
    const float4 n = *(const float4*)(zn32 + (size_t)ns  * DIMS + lane * 4);

    float sap = 0.f, san = 0.f, d;
    d = a.x - p.x + PD_EPS; sap += d * d;
    d = a.y - p.y + PD_EPS; sap += d * d;
    d = a.z - p.z + PD_EPS; sap += d * d;
    d = a.w - p.w + PD_EPS; sap += d * d;
    d = a.x - n.x + PD_EPS; san += d * d;
    d = a.y - n.y + PD_EPS; san += d * d;
    d = a.z - n.z + PD_EPS; san += d * d;
    d = a.w - n.w + PD_EPS; san += d * d;
#pragma unroll
    for (int m = 1; m < 32; m <<= 1) {
        sap += __shfl_xor(sap, m, 32);
        san += __shfl_xor(san, m, 32);
    }

    if (lane == 0) {
        const float per = fmaxf(sqrtf(sap) - sqrtf(san) + MARGIN, 0.0f);
        per_row[row] = valid ? per : 0.0f;
    }
}

// ---------------------------------------------------------------------------
// K4: deterministic fixed-order final reduction -> scalar loss.
// ---------------------------------------------------------------------------
__global__ __launch_bounds__(256)
void tl_finalize_kernel(const float* __restrict__ per_row,
                        const int* __restrict__ pidx,
                        const int* __restrict__ nidx,
                        float* __restrict__ out)
{
    __shared__ float ssum[256];
    __shared__ int   scnt[256];
    const int tid = threadIdx.x;
    float s = 0.0f;
    int   c = 0;
    for (int i = tid; i < N_ROWS; i += 256) {
        s += per_row[i];
        c += (pidx[i] >= 0 && nidx[i] >= 0) ? 1 : 0;
    }
    ssum[tid] = s; scnt[tid] = c;
    __syncthreads();
    for (int st = 128; st > 0; st >>= 1) {
        if (tid < st) { ssum[tid] += ssum[tid + st]; scnt[tid] += scnt[tid + st]; }
        __syncthreads();
    }
    if (tid == 0) {
        const int cc = scnt[0];
        out[0] = (cc > 0) ? (ssum[0] / (float)cc) : 0.0f;
    }
}

// ---------------------------------------------------------------------------
// Launch. d_ws layout (bytes):
//   [0, 2MB)      zn_f16
//   [2MB, 6MB)    zn_f32
//   then key / pidx / nidx / per_row, 32 KB each (~6.4 MB total).
// ---------------------------------------------------------------------------
extern "C" void kernel_launch(void* const* d_in, const int* in_sizes, int n_in,
                              void* d_out, int out_size, void* d_ws, size_t ws_size,
                              hipStream_t stream)
{
    (void)in_sizes; (void)n_in; (void)out_size; (void)ws_size;

    const float* z        = (const float*)d_in[0];
    const int*   labels   = (const int*)d_in[1];
    const int*   subjects = (const int*)d_in[2];

    char* ws = (char*)d_ws;
    _Float16* zn16 = (_Float16*)(ws);
    float*    zn32 = (float*)(ws + 2097152);
    int*      key  = (int*)(ws + 2097152 + 4194304);
    int*      pidx = (int*)(ws + 2097152 + 4194304 + 32768);
    int*      nidx = (int*)(ws + 2097152 + 4194304 + 65536);
    float*    per  = (float*)(ws + 2097152 + 4194304 + 98304);

    tl_normalize_kernel<<<N_ROWS / 8, 256, 0, stream>>>(z, labels, subjects,
                                                        zn32, zn16, key);
    tl_mine_kernel<<<N_ROWS / 128, 256, 0, stream>>>(zn16, key, pidx, nidx);
    tl_loss_kernel<<<N_ROWS / 8, 256, 0, stream>>>(zn32, pidx, nidx, per);
    tl_finalize_kernel<<<1, 256, 0, stream>>>(per, pidx, nidx, (float*)d_out);
}
